// BeamformingModule_46093589020859
// MI455X (gfx1250) — matrix operand order repaired and checked by reading, not compiled, not run on hardware
//
#include <hip/hip_runtime.h>
#include <hip/hip_bf16.h>
#include <math.h>

// ---------------------------------------------------------------------------
// CDNA5 (gfx1250) WMMA types & helpers
// ---------------------------------------------------------------------------
typedef __attribute__((ext_vector_type(16))) __bf16 v16bf;
typedef __attribute__((ext_vector_type(8)))  float  v8f;

union BF16x16 { v16bf v; unsigned u[8]; };

// one-time (weight prep) conversion with RNE rounding
static __device__ __forceinline__ __bf16 f2bf(float f) {
    unsigned u = __builtin_bit_cast(unsigned, f);
    unsigned r = u + 0x7FFFu + ((u >> 16) & 1u);
    unsigned short h = (unsigned short)(r >> 16);
    return __builtin_bit_cast(__bf16, h);
}

// hot path: pack two f32 -> dword of two bf16 (truncate) via v_perm_b32
static __device__ __forceinline__ unsigned pack_bf2(float a, float b) {
    unsigned ua = __builtin_bit_cast(unsigned, a);
    unsigned ub = __builtin_bit_cast(unsigned, b);
    return __builtin_amdgcn_perm(ub, ua, 0x07060302u);  // {b.hi16, a.hi16}
}

static __device__ __forceinline__ v8f wmma_bf16(v16bf a, v16bf b, v8f c) {
    return __builtin_amdgcn_wmma_f32_16x16x32_bf16(
        false, a, false, b, (short)0, c, false, false);
}

// A 16x32 (16-bit) element->K map: two contiguous 8-runs at lh*8 and 16+lh*8.
// B 32x16 (16-bit): same map as A, two 8-runs at lh*8 and 16+lh*8.

#define NEGV (-1000000000.0f)

// pack 4 float4 (16 consecutive-in-K floats, two 8-runs) into a bf16 fragment
static __device__ __forceinline__ void pack_frag(BF16x16& t, float4 a0, float4 a1,
                                                 float4 a2, float4 a3) {
    t.u[0] = pack_bf2(a0.x, a0.y); t.u[1] = pack_bf2(a0.z, a0.w);
    t.u[2] = pack_bf2(a1.x, a1.y); t.u[3] = pack_bf2(a1.z, a1.w);
    t.u[4] = pack_bf2(a2.x, a2.y); t.u[5] = pack_bf2(a2.z, a2.w);
    t.u[6] = pack_bf2(a3.x, a3.y); t.u[7] = pack_bf2(a3.z, a3.w);
}
static __device__ __forceinline__ void neg_frag(BF16x16& d, const BF16x16& s) {
    #pragma unroll
    for (int p = 0; p < 8; ++p) d.u[p] = s.u[p] ^ 0x80008000u;
}

// ---------------------------------------------------------------------------
// Weight prep: W[K][N] f32 (wr, wi) -> bf16 planes [2][Np][Kp], zero padded
// ---------------------------------------------------------------------------
__global__ void k_prep_w(const float* __restrict__ wr, const float* __restrict__ wi,
                         __bf16* __restrict__ out, int K, int N, int Kp, int Np) {
    int idx = blockIdx.x * 256 + threadIdx.x;
    if (idx >= Np * Kp) return;
    int n = idx / Kp, k = idx - n * Kp;
    bool ok = (n < N) && (k < K);
    float vr = ok ? wr[(size_t)k * N + n] : 0.0f;
    float vi = ok ? wi[(size_t)k * N + n] : 0.0f;
    out[idx]                   = f2bf(vr);
    out[(size_t)Np * Kp + idx] = f2bf(vi);
}

// rows x K  ->  rows x Kp (zero padded)
__global__ void k_padrows(const float* __restrict__ in, float* __restrict__ out,
                          int rows, int K, int Kp) {
    size_t n = (size_t)rows * Kp;
    size_t idx = (size_t)blockIdx.x * 256 + threadIdx.x;
    if (idx >= n) return;
    int r = idx / Kp, k = idx - (size_t)r * Kp;
    out[idx] = (k < K) ? in[(size_t)r * K + k] : 0.0f;
}

// ---------------------------------------------------------------------------
// Complex GEMM via bf16 WMMA, f32 accumulate. Requirements:
//   M % 64 == 0, K % 32 == 0 (padded), weights padded to Np cols.
//   Yr = Xr*Wr - Xi*Wi + br ;  Yi = Xr*Wi + Xi*Wr + bi ; optional leaky relu.
// Block = 128 (4 waves); wave computes 16 rows x 32 cols. Grid x covers
// ceil(N/32) tiles; stores/bias guarded by logical N, output stride Ns.
// ---------------------------------------------------------------------------
__global__ void k_cgemm(const float* __restrict__ Xr, const float* __restrict__ Xi,
                        const __bf16* __restrict__ Wt,   // [2][Np][K]
                        const float* __restrict__ br, const float* __restrict__ bi,
                        float* __restrict__ Yr, float* __restrict__ Yi,
                        int M, int K, int N, int Ns, int Np, int act) {
    const int lane = threadIdx.x & 31;
    const int wave = threadIdx.x >> 5;
    const int lh = lane >> 4, lm = lane & 15;
    const int m0 = blockIdx.y * 64 + wave * 16;
    const int n0 = blockIdx.x * 32;
    const int rowA = m0 + lm;
    const int col0 = n0 + lm;
    const int col1 = n0 + 16 + lm;

    const float* xr = Xr + (size_t)rowA * K;
    const float* xi = Xi + (size_t)rowA * K;
    // bf16 weight columns as dword streams (16 bf16 = 8 dwords per k-step half)
    const unsigned* bR0 = (const unsigned*)(Wt + (size_t)col0 * K);
    const unsigned* bR1 = (const unsigned*)(Wt + (size_t)col1 * K);
    const unsigned* bI0 = (const unsigned*)(Wt + ((size_t)Np + col0) * K);
    const unsigned* bI1 = (const unsigned*)(Wt + ((size_t)Np + col1) * K);

    v8f yr0 = {}, yr1 = {}, yi0 = {}, yi1 = {};

    for (int kk = 0; kk < K; kk += 32) {
        __builtin_prefetch(xr + kk + 256, 0, 3);
        __builtin_prefetch(xi + kk + 256, 0, 3);
        const float* ar_p = xr + kk + lh * 8;
        const float* ai_p = xi + kk + lh * 8;
        float4 r0 = *(const float4*)(ar_p);
        float4 r1 = *(const float4*)(ar_p + 4);
        float4 r2 = *(const float4*)(ar_p + 16);
        float4 r3 = *(const float4*)(ar_p + 20);
        float4 c0 = *(const float4*)(ai_p);
        float4 c1 = *(const float4*)(ai_p + 4);
        float4 c2 = *(const float4*)(ai_p + 16);
        float4 c3 = *(const float4*)(ai_p + 20);
        BF16x16 ar, ai, an;
        pack_frag(ar, r0, r1, r2, r3);
        pack_frag(ai, c0, c1, c2, c3);
        neg_frag(an, ai);

        const int kd = (kk + lh * 8) >> 1;        // dword offset in column
        uint4 wr0a = *(const uint4*)(bR0 + kd);
        uint4 wr0b = *(const uint4*)(bR0 + kd + 8);
        uint4 wi0a = *(const uint4*)(bI0 + kd);
        uint4 wi0b = *(const uint4*)(bI0 + kd + 8);
        uint4 wr1a = *(const uint4*)(bR1 + kd);
        uint4 wr1b = *(const uint4*)(bR1 + kd + 8);
        uint4 wi1a = *(const uint4*)(bI1 + kd);
        uint4 wi1b = *(const uint4*)(bI1 + kd + 8);
        BF16x16 br0, bi0, br1, bi1;
        br0.u[0] = wr0a.x; br0.u[1] = wr0a.y; br0.u[2] = wr0a.z; br0.u[3] = wr0a.w;
        br0.u[4] = wr0b.x; br0.u[5] = wr0b.y; br0.u[6] = wr0b.z; br0.u[7] = wr0b.w;
        bi0.u[0] = wi0a.x; bi0.u[1] = wi0a.y; bi0.u[2] = wi0a.z; bi0.u[3] = wi0a.w;
        bi0.u[4] = wi0b.x; bi0.u[5] = wi0b.y; bi0.u[6] = wi0b.z; bi0.u[7] = wi0b.w;
        br1.u[0] = wr1a.x; br1.u[1] = wr1a.y; br1.u[2] = wr1a.z; br1.u[3] = wr1a.w;
        br1.u[4] = wr1b.x; br1.u[5] = wr1b.y; br1.u[6] = wr1b.z; br1.u[7] = wr1b.w;
        bi1.u[0] = wi1a.x; bi1.u[1] = wi1a.y; bi1.u[2] = wi1a.z; bi1.u[3] = wi1a.w;
        bi1.u[4] = wi1b.x; bi1.u[5] = wi1b.y; bi1.u[6] = wi1b.z; bi1.u[7] = wi1b.w;

        yr0 = wmma_bf16(ar.v, br0.v, yr0);  yr0 = wmma_bf16(an.v, bi0.v, yr0);
        yi0 = wmma_bf16(ar.v, bi0.v, yi0);  yi0 = wmma_bf16(ai.v, br0.v, yi0);
        yr1 = wmma_bf16(ar.v, br1.v, yr1);  yr1 = wmma_bf16(an.v, bi1.v, yr1);
        yi1 = wmma_bf16(ar.v, bi1.v, yi1);  yi1 = wmma_bf16(ai.v, br1.v, yi1);
    }

    #pragma unroll
    for (int v = 0; v < 8; ++v) {
        int row = m0 + lh * 8 + v;
        size_t rb = (size_t)row * Ns;
        if (col0 < N) {
            float r = yr0[v] + br[col0], q = yi0[v] + bi[col0];
            if (act) { r = r > 0.f ? r : 0.01f * r; q = q > 0.f ? q : 0.01f * q; }
            Yr[rb + col0] = r; Yi[rb + col0] = q;
        }
        if (col1 < N) {
            float r = yr1[v] + br[col1], q = yi1[v] + bi[col1];
            if (act) { r = r > 0.f ? r : 0.01f * r; q = q > 0.f ? q : 0.01f * q; }
            Yr[rb + col1] = r; Yi[rb + col1] = q;
        }
    }
}

// ---------------------------------------------------------------------------
// Complex layer norm over last dim (A == 256). One wave per row, 4 waves/blk.
// ---------------------------------------------------------------------------
__global__ void k_cln(const float* __restrict__ xr, const float* __restrict__ xi,
                      float* __restrict__ outr, float* __restrict__ outi, int rows) {
    const int A = 256;
    int row = blockIdx.x * 4 + (threadIdx.x >> 5);
    int lane = threadIdx.x & 31;
    if (row >= rows) return;
    for (int p = 0; p < 2; ++p) {
        const float* x = (p ? xi : xr) + (size_t)row * A;
        float* o = (p ? outi : outr) + (size_t)row * A;
        float v[8]; float s = 0.f;
        #pragma unroll
        for (int i = 0; i < 8; ++i) { v[i] = x[lane * 8 + i]; s += v[i]; }
        #pragma unroll
        for (int off = 16; off > 0; off >>= 1) s += __shfl_xor(s, off, 32);
        float mean = s * (1.0f / A);
        float q = 0.f;
        #pragma unroll
        for (int i = 0; i < 8; ++i) { float d = v[i] - mean; q += d * d; }
        #pragma unroll
        for (int off = 16; off > 0; off >>= 1) q += __shfl_xor(q, off, 32);
        float inv = rsqrtf(q * (1.0f / A) + 1e-5f);
        #pragma unroll
        for (int i = 0; i < 8; ++i) o[lane * 8 + i] = (v[i] - mean) * inv;
    }
}

// ---------------------------------------------------------------------------
// 2D positional encoding + scale:  x = x*sqrt(A) + pe(c,t,f)   (A=256, D=128)
// ---------------------------------------------------------------------------
__global__ void k_posenc(float* __restrict__ xr, float* __restrict__ xi,
                         int Bb, int Cc, int Tt, int A) {
    size_t n = (size_t)Bb * Cc * Tt * A;
    size_t idx = (size_t)blockIdx.x * 256 + threadIdx.x;
    if (idx >= n) return;
    int f = idx % A;
    int t = (idx / A) % Tt;
    int c = (idx / ((size_t)A * Tt)) % Cc;
    int D = A / 2;
    float pe;
    if (f < D) {
        int m = f >> 1;
        float div = __expf(-logf(10000.f) * (2.f * m) / D);
        float ang = t * div;
        pe = (f & 1) ? __cosf(ang) : __sinf(ang);
    } else {
        int g = f - D; int m = g >> 1;
        float div = __expf(-logf(10000.f) * (2.f * m) / D);
        float ang = c * div;
        pe = (g & 1) ? __cosf(ang) : __sinf(ang);
    }
    float s = sqrtf((float)A);
    xr[idx] = xr[idx] * s + pe;
    xi[idx] = xi[idx] * s + pe;
}

// (B,D1,D2,F) -> (B,D2,D1,F)
__global__ void k_tr(const float* __restrict__ in, float* __restrict__ out,
                     int Bb, int D1, int D2, int F) {
    size_t n = (size_t)Bb * D1 * D2 * F;
    size_t idx = (size_t)blockIdx.x * 256 + threadIdx.x;
    if (idx >= n) return;
    int f = idx % F;
    int d2 = (idx / F) % D2;
    int d1 = (idx / ((size_t)F * D2)) % D1;
    int b = idx / ((size_t)F * D2 * D1);
    out[(((size_t)b * D2 + d2) * D1 + d1) * F + f] = in[idx];
}

__global__ void k_add(float* __restrict__ y, const float* __restrict__ a, size_t n) {
    size_t idx = (size_t)blockIdx.x * 256 + threadIdx.x;
    if (idx < n) y[idx] += a[idx];
}

// ---------------------------------------------------------------------------
// Channel attention (S = C = 6 -> VALU). One block(64) per (b, e=time, head).
// Layout (B, C, T, A).
// ---------------------------------------------------------------------------
__global__ void k_cattn(const float* __restrict__ qr, const float* __restrict__ qi,
                        const float* __restrict__ kr, const float* __restrict__ ki,
                        const float* __restrict__ vr, const float* __restrict__ vi,
                        const unsigned char* __restrict__ cmask,
                        float* __restrict__ outr, float* __restrict__ outi,
                        int Bb, int Cc, int Tt, int A, int H) {
    __shared__ float sr[6][6], si[6][6];
    int blk = blockIdx.x;
    int h = blk % H; blk /= H;
    int e = blk % Tt; blk /= Tt;
    int b = blk;
    int dh = A / H;
    int tid = threadIdx.x;
    size_t base = ((size_t)b * Cc * Tt + e) * A + h * dh;
    size_t sstr = (size_t)Tt * A;

    if (tid < Cc * Cc) {
        int s = tid / Cc, t = tid % Cc;
        const float* pqr = qr + base + s * sstr;
        const float* pqi = qi + base + s * sstr;
        const float* pkr = kr + base + t * sstr;
        const float* pki = ki + base + t * sstr;
        float a0 = 0, a1 = 0, a2 = 0, a3 = 0;
        for (int d = 0; d < dh; ++d) {
            a0 += pqr[d] * pkr[d]; a1 += pqi[d] * pki[d];
            a2 += pqr[d] * pki[d]; a3 += pqi[d] * pkr[d];
        }
        float sc = rsqrtf((float)dh);
        float wr_ = (a0 - a1) * sc, wi_ = (a2 + a3) * sc;
        if (!cmask[b * Cc + t]) { wr_ = NEGV; wi_ = NEGV; }
        sr[s][t] = wr_; si[s][t] = wi_;
    }
    __syncthreads();
    if (tid < 2 * Cc) {
        float (*S)[6] = (tid < Cc) ? sr : si;
        int s = tid % Cc;
        float mx = -INFINITY;
        for (int t = 0; t < Cc; ++t) mx = fmaxf(mx, S[s][t]);
        float sum = 0;
        for (int t = 0; t < Cc; ++t) { float ex = __expf(S[s][t] - mx); S[s][t] = ex; sum += ex; }
        float inv = 1.f / sum;
        for (int t = 0; t < Cc; ++t) S[s][t] *= inv;
    }
    __syncthreads();
    if (tid < dh) {
        for (int s = 0; s < Cc; ++s) {
            float cr = 0, ci = 0;
            for (int t = 0; t < Cc; ++t) {
                float ar_ = sr[s][t], ai_ = si[s][t];
                float vvr = vr[base + t * sstr + tid], vvi = vi[base + t * sstr + tid];
                cr += ar_ * vvr - ai_ * vvi;
                ci += ar_ * vvi + ai_ * vvr;
            }
            outr[base + s * sstr + tid] = cr;
            outi[base + s * sstr + tid] = ci;
        }
    }
}

// ---------------------------------------------------------------------------
// Time attention via WMMA. One block(128 = 4 waves) per (b, c, h, 16-q-tile).
// Layout (B, T, C, A). Scores (16 x T, real+imag) staged in dynamic LDS.
// ---------------------------------------------------------------------------
__global__ void k_tattn(const float* __restrict__ qr, const float* __restrict__ qi,
                        const float* __restrict__ kr, const float* __restrict__ ki,
                        const float* __restrict__ vr, const float* __restrict__ vi,
                        const unsigned char* __restrict__ tmask,
                        float* __restrict__ outr, float* __restrict__ outi,
                        int Bb, int Tt, int Cc, int A, int H) {
    extern __shared__ float sm[];           // [2][16][Tt]
    float* Sr = sm;
    float* Si = sm + 16 * Tt;
    const int dh = A / H;                   // 64
    int blk = blockIdx.x;
    int qt = blk & 31; blk >>= 5;           // Tt/16 == 32
    int h = blk % H; blk /= H;
    int c = blk % Cc; blk /= Cc;
    int b = blk;
    const int lane = threadIdx.x & 31, wave = threadIdx.x >> 5;
    const int lh = lane >> 4, lm = lane & 15;
    const size_t rstr = (size_t)Cc * A;
    const size_t base = ((size_t)b * Tt * Cc + c) * A + h * dh;

    // --- Q fragments (shared by all key tiles), ksteps over d: 0..31, 32..63
    BF16x16 aqr[2], aqi[2], aqn[2];
    {
        int qrow = qt * 16 + lm;
        const float* pr = qr + base + (size_t)qrow * rstr;
        const float* pi = qi + base + (size_t)qrow * rstr;
        #pragma unroll
        for (int ks = 0; ks < 2; ++ks) {
            const float* p0 = pr + ks * 32 + lh * 8;
            const float* p1 = pi + ks * 32 + lh * 8;
            pack_frag(aqr[ks], *(const float4*)(p0), *(const float4*)(p0 + 4),
                               *(const float4*)(p0 + 16), *(const float4*)(p0 + 20));
            pack_frag(aqi[ks], *(const float4*)(p1), *(const float4*)(p1 + 4),
                               *(const float4*)(p1 + 16), *(const float4*)(p1 + 20));
            neg_frag(aqn[ks], aqi[ks]);
        }
    }
    // --- scores: wave handles 8 key tiles
    float sc = rsqrtf((float)dh);
    for (int kt = wave * 8; kt < wave * 8 + 8; ++kt) {
        v8f vsr = {}, vsi = {};
        int key = kt * 16 + lm;
        const float* pkr = kr + base + (size_t)key * rstr;
        const float* pki = ki + base + (size_t)key * rstr;
        #pragma unroll
        for (int ks = 0; ks < 2; ++ks) {
            const float* p0 = pkr + ks * 32 + lh * 8;
            const float* p1 = pki + ks * 32 + lh * 8;
            BF16x16 bkr, bki;
            pack_frag(bkr, *(const float4*)(p0), *(const float4*)(p0 + 4),
                           *(const float4*)(p0 + 16), *(const float4*)(p0 + 20));
            pack_frag(bki, *(const float4*)(p1), *(const float4*)(p1 + 4),
                           *(const float4*)(p1 + 16), *(const float4*)(p1 + 20));
            vsr = wmma_bf16(aqr[ks].v, bkr.v, vsr); vsr = wmma_bf16(aqn[ks].v, bki.v, vsr);
            vsi = wmma_bf16(aqr[ks].v, bki.v, vsi); vsi = wmma_bf16(aqi[ks].v, bkr.v, vsi);
        }
        bool mok = tmask[b * Tt + key] != 0;
        #pragma unroll
        for (int v = 0; v < 8; ++v) {
            int qrl = lh * 8 + v;
            Sr[qrl * Tt + key] = mok ? vsr[v] * sc : NEGV;
            Si[qrl * Tt + key] = mok ? vsi[v] * sc : NEGV;
        }
    }
    __syncthreads();
    // --- softmax over keys (32 rows: 16 real + 16 imag), one thread per row
    if (threadIdx.x < 32) {
        float* S = (threadIdx.x < 16) ? Sr : Si;
        float* row = S + (threadIdx.x & 15) * Tt;
        float mx = -INFINITY;
        for (int t = 0; t < Tt; ++t) mx = fmaxf(mx, row[t]);
        float sum = 0.f;
        for (int t = 0; t < Tt; ++t) { float e = __expf(row[t] - mx); row[t] = e; sum += e; }
        float inv = 1.f / sum;
        for (int t = 0; t < Tt; ++t) row[t] *= inv;
    }
    __syncthreads();
    // --- apply: wave handles one 16-wide d tile
    {
        v8f cr = {}, ci = {};
        int dcol = wave * 16 + lm;
        const float* pvr = vr + base + dcol;
        const float* pvi = vi + base + dcol;
        for (int ks = 0; ks < Tt / 32; ++ks) {
            const float* sra = Sr + lm * Tt + ks * 32 + lh * 8;
            const float* sia = Si + lm * Tt + ks * 32 + lh * 8;
            BF16x16 ar, ai, an;
            pack_frag(ar, *(const float4*)(sra), *(const float4*)(sra + 4),
                          *(const float4*)(sra + 16), *(const float4*)(sra + 20));
            pack_frag(ai, *(const float4*)(sia), *(const float4*)(sia + 4),
                          *(const float4*)(sia + 16), *(const float4*)(sia + 20));
            neg_frag(an, ai);
            BF16x16 bvr, bvi;
            #pragma unroll
            for (int p = 0; p < 8; ++p) {
                size_t k0 = (size_t)(ks * 32 + lh * 8 + 2 * p + (p >= 4 ? 8 : 0)) * rstr;
                bvr.u[p] = pack_bf2(pvr[k0], pvr[k0 + rstr]);
                bvi.u[p] = pack_bf2(pvi[k0], pvi[k0 + rstr]);
            }
            cr = wmma_bf16(ar.v, bvr.v, cr);  cr = wmma_bf16(an.v, bvi.v, cr);
            ci = wmma_bf16(ar.v, bvi.v, ci);  ci = wmma_bf16(ai.v, bvr.v, ci);
        }
        #pragma unroll
        for (int v = 0; v < 8; ++v) {
            int qrow = qt * 16 + lh * 8 + v;
            size_t o = ((size_t)b * Tt + qrow) * rstr + (size_t)c * A + h * dh + wave * 16 + lm;
            outr[o] = cr[v];
            outi[o] = ci[v];
        }
    }
}

// ---------------------------------------------------------------------------
// Final combine + mean over channel dim (Or/Oi padded stride Fp):
//   mr = or*xcr + oi*xci ; mi = or*xci - oi*xcr ; mean over C
// d_out layout: [mr (B,T,F)] ++ [mi (B,T,F)]
// ---------------------------------------------------------------------------
__global__ void k_final(const float* __restrict__ Or, const float* __restrict__ Oi,
                        const float* __restrict__ xcr, const float* __restrict__ xci,
                        float* __restrict__ out, int Bb, int Cc, int Tt, int F, int Fp) {
    size_t n = (size_t)Bb * Tt * F;
    size_t idx = (size_t)blockIdx.x * 256 + threadIdx.x;
    if (idx >= n) return;
    int f = idx % F;
    int t = (idx / F) % Tt;
    int b = idx / ((size_t)F * Tt);
    float mr = 0.f, mi = 0.f;
    for (int c = 0; c < Cc; ++c) {
        size_t rowi = ((size_t)b * Cc + c) * Tt + t;
        float ov = Or[rowi * Fp + f], ow = Oi[rowi * Fp + f];
        float xr = xcr[rowi * F + f], xi = xci[rowi * F + f];
        mr += ov * xr + ow * xi;
        mi += ov * xi - ow * xr;
    }
    float inv = 1.0f / Cc;
    out[idx] = mr * inv;
    out[n + idx] = mi * inv;
}

// ---------------------------------------------------------------------------
// Host orchestration
// ---------------------------------------------------------------------------
extern "C" void kernel_launch(void* const* d_in, const int* in_sizes, int n_in,
                              void* d_out, int out_size, void* d_ws, size_t ws_size,
                              hipStream_t stream) {
    (void)in_sizes; (void)n_in; (void)out_size; (void)ws_size;
    const int Bb = 4, Cc = 6, Tt = 512, FIN = 257, A = 256, FFN = 2048, H = 4, NL = 2;
    const int FINp = 288;                          // FIN padded to 32
    const int NTOK = Bb * Cc * Tt;                 // 12288
    const size_t ACT = (size_t)NTOK * A;

    const float* xcr = (const float*)d_in[0];
    const float* xci = (const float*)d_in[1];
    const unsigned char* cmask = (const unsigned char*)d_in[2];
    const float* xtr = (const float*)d_in[3];
    const float* xti = (const float*)d_in[4];
    const unsigned char* tmask = (const unsigned char*)d_in[5];
    auto P = [&](int i) { return (const float*)d_in[i]; };
    // pytree (sorted-key) flat order: params start at 6.
    // ch_in: 6..9 {bi,br,wi,wr}; layers l: base=10+40l
    //   ch_att {k:+0,o:+4,q:+8,v:+12}, ff1:+16, ff2:+20, t_att {k:+24,o:+28,q:+32,v:+36}
    // out: 90..93 ; t_in: 94..97. Each linear: {bi,br,wi,wr}.

    // ---- workspace carving
    char* W = (char*)d_ws; size_t off = 0;
    auto allocf = [&](size_t nelem) {
        float* p = (float*)(W + off); off += ((nelem * 4 + 255) & ~(size_t)255); return p;
    };
    auto allocb = [&](size_t nelem) {
        __bf16* p = (__bf16*)(W + off); off += ((nelem * 2 + 255) & ~(size_t)255); return p;
    };
    float *xr = allocf(ACT), *xi = allocf(ACT);          // activations (B,C,T,A)
    float *txr = allocf(ACT), *txi = allocf(ACT);        // (B,T,C,A)
    float *tfr = allocf(ACT), *tfi = allocf(ACT);        // t features (B,T,C,A)
    float *nr = allocf(ACT), *ni = allocf(ACT);
    float *qr_ = allocf(ACT), *qi_ = allocf(ACT);
    float *kr_ = allocf(ACT), *ki_ = allocf(ACT);
    float *vr_ = allocf(ACT), *vi_ = allocf(ACT);
    float *atr = allocf(ACT), *ati = allocf(ACT);
    float *pr = allocf(ACT), *pi = allocf(ACT);
    const int MCH = 2048;                                 // FFN row chunk
    float *hr = allocf((size_t)MCH * FFN), *hi = allocf((size_t)MCH * FFN);
    float *our = allocf((size_t)NTOK * FINp), *oui = allocf((size_t)NTOK * FINp);
    float *xpcr = allocf((size_t)NTOK * FINp), *xpci = allocf((size_t)NTOK * FINp);
    float *xptr_ = allocf((size_t)NTOK * FINp), *xpti = allocf((size_t)NTOK * FINp);

    __bf16* w_chin = allocb(2 * (size_t)A * FINp);
    __bf16* w_tin  = allocb(2 * (size_t)A * FINp);
    __bf16* w_out  = allocb(2 * (size_t)FINp * A);
    __bf16 *w_qc[NL], *w_kc[NL], *w_vc[NL], *w_oc[NL];
    __bf16 *w_qt[NL], *w_kt[NL], *w_vt[NL], *w_ot[NL];
    __bf16 *w_f1[NL], *w_f2[NL];
    for (int l = 0; l < NL; ++l) {
        w_kc[l] = allocb(2 * (size_t)A * A); w_oc[l] = allocb(2 * (size_t)A * A);
        w_qc[l] = allocb(2 * (size_t)A * A); w_vc[l] = allocb(2 * (size_t)A * A);
        w_kt[l] = allocb(2 * (size_t)A * A); w_ot[l] = allocb(2 * (size_t)A * A);
        w_qt[l] = allocb(2 * (size_t)A * A); w_vt[l] = allocb(2 * (size_t)A * A);
        w_f1[l] = allocb(2 * (size_t)FFN * A); w_f2[l] = allocb(2 * (size_t)A * FFN);
    }

    auto prep = [&](int pbase, __bf16* dst, int K, int N, int Kp, int Np) {
        int ne = Np * Kp;
        k_prep_w<<<(ne + 255) / 256, 256, 0, stream>>>(P(pbase + 3), P(pbase + 2),
                                                       dst, K, N, Kp, Np);
    };
    prep(6,  w_chin, FIN, A, FINp, A);
    prep(94, w_tin,  FIN, A, FINp, A);
    prep(90, w_out,  A, FIN, A, FINp);
    for (int l = 0; l < NL; ++l) {
        int b0 = 10 + 40 * l;
        prep(b0 + 0,  w_kc[l], A, A, A, A); prep(b0 + 4,  w_oc[l], A, A, A, A);
        prep(b0 + 8,  w_qc[l], A, A, A, A); prep(b0 + 12, w_vc[l], A, A, A, A);
        prep(b0 + 16, w_f1[l], A, FFN, A, FFN); prep(b0 + 20, w_f2[l], FFN, A, FFN, A);
        prep(b0 + 24, w_kt[l], A, A, A, A); prep(b0 + 28, w_ot[l], A, A, A, A);
        prep(b0 + 32, w_qt[l], A, A, A, A); prep(b0 + 36, w_vt[l], A, A, A, A);
    }
    // pad the FIN=257 inputs to 288 columns
    {
        size_t np = (size_t)NTOK * FINp;
        int g = (int)((np + 255) / 256);
        k_padrows<<<g, 256, 0, stream>>>(xcr, xpcr, NTOK, FIN, FINp);
        k_padrows<<<g, 256, 0, stream>>>(xci, xpci, NTOK, FIN, FINp);
        k_padrows<<<g, 256, 0, stream>>>(xtr, xptr_, NTOK, FIN, FINp);
        k_padrows<<<g, 256, 0, stream>>>(xti, xpti, NTOK, FIN, FINp);
    }

    auto cgemm = [&](const float* Xr, const float* Xi, const __bf16* Wt,
                     int pbase, float* Yr, float* Yi,
                     int M, int K, int N, int Ns, int Np, int act) {
        dim3 g((N + 31) / 32, M / 64);
        k_cgemm<<<g, 128, 0, stream>>>(Xr, Xi, Wt, P(pbase + 1), P(pbase + 0),
                                       Yr, Yi, M, K, N, Ns, Np, act);
    };
    size_t nAct = ACT;
    auto add = [&](float* y, const float* a) {
        k_add<<<(int)((nAct + 255) / 256), 256, 0, stream>>>(y, a, nAct);
    };
    auto cln = [&](const float* a, const float* b, float* c, float* d) {
        k_cln<<<NTOK / 4, 128, 0, stream>>>(a, b, c, d, NTOK);
    };
    auto tr = [&](const float* a, float* b, int D1, int D2) {
        k_tr<<<(int)((nAct + 255) / 256), 256, 0, stream>>>(a, b, Bb, D1, D2, A);
    };

    // ---- input projections + positional encodings
    cgemm(xpcr, xpci, w_chin, 6, xr, xi, NTOK, FINp, A, A, A, 0);
    k_posenc<<<(int)((nAct + 255) / 256), 256, 0, stream>>>(xr, xi, Bb, Cc, Tt, A);

    cgemm(xptr_, xpti, w_tin, 94, pr, pi, NTOK, FINp, A, A, A, 0);
    k_posenc<<<(int)((nAct + 255) / 256), 256, 0, stream>>>(pr, pi, Bb, Cc, Tt, A);
    cln(pr, pi, nr, ni);
    tr(nr, tfr, Cc, Tt); tr(ni, tfi, Cc, Tt);

    // ---- layers
    for (int l = 0; l < NL; ++l) {
        int b0 = 10 + 40 * l;
        // channel attention (B,C,T,A)
        cln(xr, xi, nr, ni);
        cgemm(nr, ni, w_qc[l], b0 + 8,  qr_, qi_, NTOK, A, A, A, A, 0);
        cgemm(nr, ni, w_kc[l], b0 + 0,  kr_, ki_, NTOK, A, A, A, A, 0);
        cgemm(nr, ni, w_vc[l], b0 + 12, vr_, vi_, NTOK, A, A, A, A, 0);
        k_cattn<<<Bb * Tt * H, 64, 0, stream>>>(qr_, qi_, kr_, ki_, vr_, vi_, cmask,
                                                atr, ati, Bb, Cc, Tt, A, H);
        cgemm(atr, ati, w_oc[l], b0 + 4, pr, pi, NTOK, A, A, A, A, 0);
        add(xr, pr); add(xi, pi);
        // transpose -> (B,T,C,A)
        tr(xr, txr, Cc, Tt); tr(xi, txi, Cc, Tt);
        // time attention
        cln(txr, txi, nr, ni);
        cgemm(tfr, tfi, w_qt[l], b0 + 32, qr_, qi_, NTOK, A, A, A, A, 0);
        cgemm(nr, ni,  w_kt[l], b0 + 24, kr_, ki_, NTOK, A, A, A, A, 0);
        cgemm(nr, ni,  w_vt[l], b0 + 36, vr_, vi_, NTOK, A, A, A, A, 0);
        k_tattn<<<Bb * Cc * H * (Tt / 16), 128, 2 * 16 * Tt * sizeof(float), stream>>>(
            qr_, qi_, kr_, ki_, vr_, vi_, tmask, atr, ati, Bb, Tt, Cc, A, H);
        cgemm(atr, ati, w_ot[l], b0 + 28, pr, pi, NTOK, A, A, A, A, 0);
        add(txr, pr); add(txi, pi);
        // FFN (row-chunked to bound the hidden buffer)
        cln(txr, txi, nr, ni);
        for (int m0 = 0; m0 < NTOK; m0 += MCH) {
            cgemm(nr + (size_t)m0 * A, ni + (size_t)m0 * A, w_f1[l], b0 + 16,
                  hr, hi, MCH, A, FFN, FFN, FFN, 1);
            cgemm(hr, hi, w_f2[l], b0 + 20,
                  pr + (size_t)m0 * A, pi + (size_t)m0 * A, MCH, FFN, A, A, A, 0);
        }
        add(txr, pr); add(txi, pi);
        // transpose back -> (B,C,T,A)
        tr(txr, xr, Tt, Cc); tr(txi, xi, Tt, Cc);
    }

    // ---- output head
    cln(xr, xi, nr, ni);
    cgemm(nr, ni, w_out, 90, our, oui, NTOK, A, FIN, FINp, FINp, 0);
    size_t nOut = (size_t)Bb * Tt * FIN;
    k_final<<<(int)((nOut + 255) / 256), 256, 0, stream>>>(our, oui, xcr, xci,
                                                           (float*)d_out, Bb, Cc, Tt, FIN, FINp);
}
